// RGAT_16252156248488
// MI455X (gfx1250) — matrix operand order repaired and checked by reading, not compiled, not run on hardware
//
#include <hip/hip_runtime.h>
#include <hip/hip_bf16.h>

// ---------------------------------------------------------------------------
// RGAT (2-layer, 2-relation, 2-head) for MI455X / gfx1250 (wave32, WMMA).
// Dense GEMMs use v_wmma_f32_16x16x32_bf16. Weights are pre-converted to bf16
// in WMMA fragment layout (one tiny prep kernel per GEMM), so the hot loop is
// contiguous b128 loads + native f32->bf16 converts feeding back-to-back WMMAs.
// Edge softmax/aggregation: 3 bandwidth-bound passes with float atomics.
// ---------------------------------------------------------------------------

typedef __bf16 v16bf __attribute__((ext_vector_type(16)));
typedef float  v8f   __attribute__((ext_vector_type(8)));

#define HEADS 2
#define FDIM  128   // H * HID == H * OUT == 128 for both layers

// native conversion: lowers to v_cvt_*bf16* on gfx1250 (true bf16 VALU arch)
__device__ __forceinline__ __bf16 f2bf(float f) { return (__bf16)f; }

__device__ __forceinline__ void atomicMaxF(float* addr, float val) {
    // signed-int max for non-negative, unsigned-int min for negative floats
    if (val >= 0.0f) atomicMax((int*)addr, __float_as_int(val));
    else             atomicMin((unsigned int*)addr, (unsigned int)__float_as_int(val));
}

// ---------------------------------------------------------------------------
// Generic fill
// ---------------------------------------------------------------------------
__global__ void fill_kernel(float* __restrict__ p, float v, long long n) {
    long long i = (long long)blockIdx.x * blockDim.x + threadIdx.x;
    if (i < n) p[i] = v;
}

// ---------------------------------------------------------------------------
// B fragment prep: convert B[K x NC] (f32, row-major) to bf16 stored in the
// per-lane WMMA fragment order:
//   Bf[ ((kc*2 + hi)*NC + col)*16 + i ] = bf16( B[(kc*32 + hi*16 + i)*NC + col] )
// so a lane's 16 K-values for one (chunk, column) are 32 contiguous bytes.
// ---------------------------------------------------------------------------
__global__ void bprep_kernel(const float* __restrict__ B, __bf16* __restrict__ Bf,
                             int K, int NC)
{
    int idx = blockIdx.x * blockDim.x + threadIdx.x;     // over K*NC
    if (idx >= K * NC) return;
    int i   = idx & 15;
    int col = (idx >> 4) % NC;
    int g   = idx / (16 * NC);                           // g = kc*2 + hi
    int kc  = g >> 1, hi = g & 1;
    int k   = kc * 32 + hi * 16 + i;
    Bf[idx] = f2bf(B[(size_t)k * NC + col]);
}

// ---------------------------------------------------------------------------
// WMMA GEMM:  Out[M x NC] = act( (A[M x K] + Abias0 + Abias1) @ B + outBias )
//   - B supplied as bf16 fragments (bprep_kernel layout)
//   - 256-thread block = 8 waves; wave w owns row strip (blockIdx.x*8+w)*16
//   - wave keeps NC/16 accumulator tiles; A fragment loaded once per K-chunk
//   - all flags compile-time -> branch-free inner loop
//   - K % 32 == 0, K <= 128; NC in {64, 128}
// ---------------------------------------------------------------------------
template <int K, int NC, bool ABIAS, bool RELU, bool OBIAS>
__global__ __launch_bounds__(256)
void gemm_wmma(const float* __restrict__ A,
               const float* __restrict__ Abias0,
               const float* __restrict__ Abias1,
               const __bf16* __restrict__ Bf,
               const float* __restrict__ outBias,
               float* __restrict__ Out, int M)
{
    constexpr int NT = NC / 16;                 // column tiles per wave
    const int lane    = threadIdx.x & 31;
    const int wave    = threadIdx.x >> 5;
    const int rowTile = blockIdx.x * 8 + wave;  // 16-row strip (wave-uniform)

    if (rowTile * 16 >= M) return;              // wave-uniform: EXEC stays all-ones below

    if (rowTile * 16 + 16 > M) {
        // wave-uniform scalar tail (no WMMA on this path)
        int r = rowTile * 16 + lane;
        if (r < M) {
            for (int col = 0; col < NC; ++col) {
                float acc = 0.0f;
                for (int k = 0; k < K; ++k) {
                    float a = A[(size_t)r * K + k];
                    if (ABIAS) a += Abias0[k] + Abias1[k];
                    int kc = k >> 5, hi = (k >> 4) & 1, i = k & 15;
                    float b = (float)Bf[((size_t)(kc * 2 + hi) * NC + col) * 16 + i];
                    acc += a * b;
                }
                if (OBIAS) acc += outBias[col];
                if (RELU)  acc = fmaxf(acc, 0.0f);
                Out[(size_t)r * NC + col] = acc;
            }
        }
        return;
    }

    const int row     = lane & 15;
    const int hi      = (lane >> 4) & 1;        // lane group 0-15 vs 16-31
    const int colLane = lane & 15;
    const size_t rowg = (size_t)rowTile * 16 + row;

    v8f c[NT];
#pragma unroll
    for (int t = 0; t < NT; ++t)
        c[t] = (v8f){0.f, 0.f, 0.f, 0.f, 0.f, 0.f, 0.f, 0.f};

#pragma unroll
    for (int kc = 0; kc < K / 32; ++kc) {
        // ---- A fragment: contiguous k order, same as the B fragments ----
        // lanes 0-15: K = kc*32 + {0..15}; lanes 16-31: +16
        const int ka = kc * 32 + hi * 16;
        const float* ap = A + rowg * K + ka;
        v16bf a;
#pragma unroll
        for (int i = 0; i < 8; ++i) {
            float v0 = ap[i];
            float v1 = ap[8 + i];
            if (ABIAS) {
                v0 += Abias0[ka + i]      + Abias1[ka + i];
                v1 += Abias0[ka + 8 + i] + Abias1[ka + 8 + i];
            }
            a[i]     = f2bf(v0);
            a[8 + i] = f2bf(v1);
        }
        // ---- B fragments (contiguous 32B bf16 loads) + WMMA, A reused ----
        const __bf16* bbase = Bf + (size_t)(kc * 2 + hi) * NC * 16;
#pragma unroll
        for (int t = 0; t < NT; ++t) {
            v16bf b = *(const v16bf*)(bbase + (size_t)(t * 16 + colLane) * 16);
            c[t] = __builtin_amdgcn_wmma_f32_16x16x32_bf16(
                       false, a, false, b, (short)0, c[t], false, false);
        }
    }

    // ---- epilogue: C/D layout VGPR j -> row j (+8 for upper lane group) ----
    const size_t rbase = (size_t)rowTile * 16 + hi * 8;
#pragma unroll
    for (int t = 0; t < NT; ++t) {
        const int col = t * 16 + colLane;
        const float ob = OBIAS ? outBias[col] : 0.0f;
#pragma unroll
        for (int j = 0; j < 8; ++j) {
            float v = c[t][j] + ob;
            if (RELU) v = fmaxf(v, 0.0f);
            Out[(rbase + j) * NC + col] = v;
        }
    }
}

// ---------------------------------------------------------------------------
// Per-node attention scores: el[n,h] = <feat[n,h,:], al[h,:]>, er likewise.
// One wave per node, shuffle reduction (wave32).
// ---------------------------------------------------------------------------
__global__ void scores_kernel(const float* __restrict__ feat,
                              const float* __restrict__ al,
                              const float* __restrict__ ar,
                              float* __restrict__ el,
                              float* __restrict__ er, int n)
{
    int gw   = (int)(((long long)blockIdx.x * blockDim.x + threadIdx.x) >> 5);
    int lane = threadIdx.x & 31;
    if (gw >= n) return;
    const float* f = feat + (size_t)gw * FDIM;
#pragma unroll
    for (int h = 0; h < HEADS; ++h) {
        float f0 = f[h * 64 + lane];
        float f1 = f[h * 64 + lane + 32];
        float pl = f0 * al[h * 64 + lane] + f1 * al[h * 64 + lane + 32];
        float pr = f0 * ar[h * 64 + lane] + f1 * ar[h * 64 + lane + 32];
#pragma unroll
        for (int off = 16; off >= 1; off >>= 1) {
            pl += __shfl_xor(pl, off, 32);
            pr += __shfl_xor(pr, off, 32);
        }
        if (lane == 0) { el[gw * HEADS + h] = pl; er[gw * HEADS + h] = pr; }
    }
}

__device__ __forceinline__ float leaky(float v) { return v > 0.0f ? v : 0.2f * v; }

// ---- pass 1: segment max over destination nodes ----
__global__ void edge_max_kernel(const int* __restrict__ src, const int* __restrict__ dst,
                                const float* __restrict__ el, const float* __restrict__ er,
                                float* __restrict__ m, int E)
{
    int e = blockIdx.x * blockDim.x + threadIdx.x;
    if (e >= E) return;
    int s = src[e], d = dst[e];
#pragma unroll
    for (int h = 0; h < HEADS; ++h) {
        float v = leaky(el[s * HEADS + h] + er[d * HEADS + h]);
        atomicMaxF(&m[d * HEADS + h], v);
    }
}

// ---- pass 2: exp-sum ----
__global__ void edge_expsum_kernel(const int* __restrict__ src, const int* __restrict__ dst,
                                   const float* __restrict__ el, const float* __restrict__ er,
                                   const float* __restrict__ m, float* __restrict__ den, int E)
{
    int e = blockIdx.x * blockDim.x + threadIdx.x;
    if (e >= E) return;
    int s = src[e], d = dst[e];
#pragma unroll
    for (int h = 0; h < HEADS; ++h) {
        float v = leaky(el[s * HEADS + h] + er[d * HEADS + h]);
        atomicAdd(&den[d * HEADS + h], __expf(v - m[d * HEADS + h]));
    }
}

// ---- pass 3: alpha-weighted scatter of features; one wave per edge ----
__global__ void edge_agg_kernel(const int* __restrict__ src, const int* __restrict__ dst,
                                const float* __restrict__ el, const float* __restrict__ er,
                                const float* __restrict__ m, const float* __restrict__ den,
                                const float* __restrict__ feat, float* __restrict__ acc, int E)
{
    int gw   = (int)(((long long)blockIdx.x * blockDim.x + threadIdx.x) >> 5);
    int lane = threadIdx.x & 31;
    if (gw >= E) return;
    int s = src[gw], d = dst[gw];
    float alpha[HEADS];
#pragma unroll
    for (int h = 0; h < HEADS; ++h) {
        float v = leaky(el[s * HEADS + h] + er[d * HEADS + h]);
        alpha[h] = __expf(v - m[d * HEADS + h]) / den[d * HEADS + h];
    }
    const float* fs = feat + (size_t)s * FDIM;
    float*       ad = acc  + (size_t)d * FDIM;
#pragma unroll
    for (int i = 0; i < FDIM / 32; ++i) {
        int f = lane + i * 32;
        atomicAdd(&ad[f], fs[f] * alpha[f >> 6]);
    }
}

// ---------------------------------------------------------------------------
// Launch
// ---------------------------------------------------------------------------
extern "C" void kernel_launch(void* const* d_in, const int* in_sizes, int n_in,
                              void* d_out, int out_size, void* d_ws, size_t ws_size,
                              hipStream_t stream)
{
    const float* x     = (const float*)d_in[0];
    const int*   src   = (const int*)  d_in[1];
    const int*   dst   = (const int*)  d_in[2];
    const float* W0    = (const float*)d_in[3];
    const float* al0   = (const float*)d_in[4];
    const float* ar0   = (const float*)d_in[5];
    const float* b0    = (const float*)d_in[6];
    const float* W1    = (const float*)d_in[7];
    const float* al1   = (const float*)d_in[8];
    const float* ar1   = (const float*)d_in[9];
    const float* b1    = (const float*)d_in[10];
    const float* linW0 = (const float*)d_in[11];
    const float* linb0 = (const float*)d_in[12];
    const float* linW1 = (const float*)d_in[13];
    const float* linb1 = (const float*)d_in[14];

    const int N = in_sizes[0] / 128;   // IN = 128
    const int E = in_sizes[1] / 2;     // R  = 2
    const int R = 2;

    // workspace layout
    float* feat = (float*)d_ws;            // N*128
    float* acc  = feat + (size_t)N * FDIM; // N*128
    float* h    = acc  + (size_t)N * FDIM; // N*128
    float* el   = h    + (size_t)N * FDIM; // N*2
    float* er   = el   + (size_t)N * HEADS;
    float* m    = er   + (size_t)N * HEADS;
    float* den  = m    + (size_t)N * HEADS;
    // bf16 fragment buffer for the current weight matrix (max 128*128), 32B aligned
    uintptr_t bfp = (uintptr_t)(den + (size_t)N * HEADS);
    bfp = (bfp + 31) & ~(uintptr_t)31;
    __bf16* bfrag = (__bf16*)bfp;

    const long long nFeat = (long long)N * FDIM;
    const long long nNH   = (long long)N * HEADS;
    const int fillFeatG = (int)((nFeat + 255) / 256);
    const int fillNHG   = (int)((nNH + 255) / 256);
    const int gemmG     = ((N + 15) / 16 + 7) / 8;   // 8 row-strips per block
    const int scoresG   = (int)(((long long)N * 32 + 255) / 256);
    const int edgeG     = (E + 255) / 256;
    const int edgeWaveG = (int)(((long long)E * 32 + 255) / 256);

    // ================= layer 0 =================
    fill_kernel<<<fillFeatG, 256, 0, stream>>>(acc, 0.0f, nFeat);
    for (int r = 0; r < R; ++r) {
        const int* sr = src + (size_t)r * E;
        const int* dr = dst + (size_t)r * E;
        bprep_kernel<<<(128 * 128 + 255) / 256, 256, 0, stream>>>(
            W0 + (size_t)r * 128 * 128, bfrag, 128, 128);
        gemm_wmma<128, 128, false, false, false><<<gemmG, 256, 0, stream>>>(
            x, nullptr, nullptr, bfrag, nullptr, feat, N);
        scores_kernel<<<scoresG, 256, 0, stream>>>(feat, al0 + r * 128, ar0 + r * 128, el, er, N);
        fill_kernel<<<fillNHG, 256, 0, stream>>>(m, -1e30f, nNH);
        fill_kernel<<<fillNHG, 256, 0, stream>>>(den, 0.0f, nNH);
        edge_max_kernel<<<edgeG, 256, 0, stream>>>(sr, dr, el, er, m, E);
        edge_expsum_kernel<<<edgeG, 256, 0, stream>>>(sr, dr, el, er, m, den, E);
        edge_agg_kernel<<<edgeWaveG, 256, 0, stream>>>(sr, dr, el, er, m, den, feat, acc, E);
    }
    // h = relu( (acc + b0[0] + b0[1]) @ linW0 + linb0 )
    bprep_kernel<<<(128 * 64 + 255) / 256, 256, 0, stream>>>(linW0, bfrag, 128, 64);
    gemm_wmma<128, 64, true, true, true><<<gemmG, 256, 0, stream>>>(
        acc, b0, b0 + 128, bfrag, linb0, h, N);

    // ================= layer 1 =================
    fill_kernel<<<fillFeatG, 256, 0, stream>>>(acc, 0.0f, nFeat);
    for (int r = 0; r < R; ++r) {
        const int* sr = src + (size_t)r * E;
        const int* dr = dst + (size_t)r * E;
        bprep_kernel<<<(64 * 128 + 255) / 256, 256, 0, stream>>>(
            W1 + (size_t)r * 64 * 128, bfrag, 64, 128);
        gemm_wmma<64, 128, false, false, false><<<gemmG, 256, 0, stream>>>(
            h, nullptr, nullptr, bfrag, nullptr, feat, N);
        scores_kernel<<<scoresG, 256, 0, stream>>>(feat, al1 + r * 128, ar1 + r * 128, el, er, N);
        fill_kernel<<<fillNHG, 256, 0, stream>>>(m, -1e30f, nNH);
        fill_kernel<<<fillNHG, 256, 0, stream>>>(den, 0.0f, nNH);
        edge_max_kernel<<<edgeG, 256, 0, stream>>>(sr, dr, el, er, m, E);
        edge_expsum_kernel<<<edgeG, 256, 0, stream>>>(sr, dr, el, er, m, den, E);
        edge_agg_kernel<<<edgeWaveG, 256, 0, stream>>>(sr, dr, el, er, m, den, feat, acc, E);
    }
    // out = (acc + b1[0] + b1[1]) @ linW1 + linb1
    bprep_kernel<<<(128 * 64 + 255) / 256, 256, 0, stream>>>(linW1, bfrag, 128, 64);
    gemm_wmma<128, 64, true, false, true><<<gemmG, 256, 0, stream>>>(
        acc, b1, b1 + 128, bfrag, linb1, (float*)d_out, N);
}
